// EncoderDecoder_7962869367385
// MI455X (gfx1250) — matrix operand order repaired and checked by reading, not compiled, not run on hardware
//
#include <hip/hip_runtime.h>
#include <hip/hip_bf16.h>
#include <math.h>

// ---------------- problem constants ----------------
#define Bb   32
#define Tt   20
#define Vv   30000
#define Hh   512
#define Ee   512
#define Aa   512
#define G3H  1536    // 3*H
#define CAT  1536    // 3*A

typedef __attribute__((ext_vector_type(16))) _Float16 v16h;
typedef __attribute__((ext_vector_type(8)))  _Float16 v8h;
typedef __attribute__((ext_vector_type(8)))  float    v8f;

__device__ __forceinline__ float sigmoidf_(float x) { return 1.0f / (1.0f + __expf(-x)); }

// ---------------------------------------------------------------------------
// f32 -> f16 converter (one-time weight conversion; halves L2/HBM stream)
// ---------------------------------------------------------------------------
__global__ void cvt_f16_kernel(const float* __restrict__ s, _Float16* __restrict__ d, long n) {
    long i = (long)blockIdx.x * blockDim.x + threadIdx.x;
    long stride = (long)gridDim.x * blockDim.x;
    for (; i < n; i += stride) d[i] = (_Float16)s[i];
}

// ---------------------------------------------------------------------------
// WMMA GEMM:  C[32 x N] = X[32 x K](f16) @ W[N x K]^T(f16)  (+bias), f32 out.
// One wave per 16x16 tile; 8 waves/block = 2 m-tiles x 4 n-tiles.
// Fragment layouts follow CDNA5 ISA 7.12.2 (16-bit A 16x32, B 32x16, f32 C/D).
// ---------------------------------------------------------------------------
__global__ void gemm_xwT_wmma(const _Float16* __restrict__ X,
                              const _Float16* __restrict__ W,
                              const float* __restrict__ bias,
                              float* __restrict__ C,
                              int K, int Ntiles) {
    const int lane  = threadIdx.x & 31;
    const int wave  = threadIdx.x >> 5;           // 0..7
    const int mtile = wave & 1;                   // B=32 -> 2 row tiles
    const int ntile = blockIdx.x * 4 + (wave >> 1);
    if (ntile >= Ntiles) return;                  // whole-wave early exit: EXEC stays all-1s
    const int N  = Ntiles * 16;
    const int hi = lane >> 4;                     // lane half
    const int ln = lane & 15;
    const int m  = mtile * 16 + ln;               // A row held by this lane
    const int n  = ntile * 16 + ln;               // B column held by this lane

    const _Float16* xrow = X + (size_t)m * K;
    const _Float16* wrow = W + (size_t)n * K;

    v8f acc = {};
    for (int k0 = 0; k0 < K; k0 += 32) {
        __builtin_prefetch((const void*)(wrow + k0 + 128), 0, 0); // global_prefetch_b8
        // A fragment: elems 0-7 -> K = k0+hi*8 .. +7 ; elems 8-15 -> K = k0+16+hi*8 ..
        v8h alo = *(const v8h*)(xrow + k0 + hi * 8);
        v8h ahi = *(const v8h*)(xrow + k0 + 16 + hi * 8);
        v16h a;
#pragma unroll
        for (int i = 0; i < 8; ++i) { a[i] = alo[i]; a[8 + i] = ahi[i]; }
        // B fragment: elems 0-7 -> K = k0+hi*8 .. +7 ; elems 8-15 -> K = k0+16+hi*8 ..
        v8h blo = *(const v8h*)(wrow + k0 + hi * 8);
        v8h bhi = *(const v8h*)(wrow + k0 + 16 + hi * 8);
        v16h b;
#pragma unroll
        for (int i = 0; i < 8; ++i) { b[i] = blo[i]; b[8 + i] = bhi[i]; }
        acc = __builtin_amdgcn_wmma_f32_16x16x32_f16(
                  false, a, false, b, (short)0, acc, false, false);
    }
    const float bn = bias ? bias[n] : 0.0f;
#pragma unroll
    for (int r = 0; r < 8; ++r) {                 // D layout: VGPR r, lanes0-15 M=r, lanes16-31 M=8+r
        const int mr = mtile * 16 + hi * 8 + r;
        C[(size_t)mr * N + n] = acc[r] + bn;
    }
}

// ---------------------------------------------------------------------------
// Encoder init: attr_cat, attr_final (-> h0,h1,ctx), proj_key.  One block / batch row.
// ---------------------------------------------------------------------------
__global__ void encoder_kernel(const int* __restrict__ user, const int* __restrict__ product,
                               const int* __restrict__ rating,
                               const float* __restrict__ user_w, const float* __restrict__ product_w,
                               const float* __restrict__ rating_w,
                               const float* __restrict__ enc_w, const float* __restrict__ enc_b,
                               const float* __restrict__ key_w,
                               float* __restrict__ attr_cat, float* __restrict__ proj_key,
                               float* __restrict__ h0, _Float16* __restrict__ h0h,
                               float* __restrict__ h1, _Float16* __restrict__ h1h,
                               float* __restrict__ ctx) {
    __shared__ float sc[CAT];
    const int b = blockIdx.x, tid = threadIdx.x;
    const int u = user[b], pr = product[b], rt = rating[b];
    for (int i = tid; i < CAT; i += 256) {
        float v = (i < Aa)     ? user_w[(size_t)u * Aa + i]
                : (i < 2 * Aa) ? product_w[(size_t)pr * Aa + (i - Aa)]
                               : rating_w[(size_t)rt * Aa + (i - 2 * Aa)];
        sc[i] = v;
        attr_cat[(size_t)b * CAT + i] = v;
    }
    __syncthreads();
    // attr_final = leaky_relu(attr_cat @ enc_w.T + enc_b) : [1024]
    for (int j = tid; j < 2 * Hh; j += 256) {
        float a = enc_b[j];
        const float* wr = enc_w + (size_t)j * CAT;
        for (int k = 0; k < CAT; ++k) a = fmaf(sc[k], wr[k], a);
        float af = (a > 0.0f) ? a : 0.01f * a;
        if (j < Hh) {
            h0[(size_t)b * Hh + j] = af; h0h[(size_t)b * Hh + j] = (_Float16)af;
        } else {
            int jj = j - Hh;
            h1[(size_t)b * Hh + jj] = af; h1h[(size_t)b * Hh + jj] = (_Float16)af;
            ctx[(size_t)b * Hh + jj] = af;      // L=2: attr_final[-1] == attr_final[1]
        }
    }
    // proj_key[b, s, h] = encoder_attr[b,s,:] . key_w[h,:]
    for (int o = tid; o < 3 * Hh; o += 256) {
        const int s = o / Hh, h = o % Hh;
        const float* cs = sc + s * Aa;
        const float* kr = key_w + (size_t)h * Aa;
        float a = 0.0f;
        for (int k = 0; k < Aa; ++k) a = fmaf(cs[k], kr[k], a);
        proj_key[(size_t)b * (3 * Hh) + o] = a;
    }
}

// ---------------------------------------------------------------------------
// Stage rnn_in (f16) = [ embed[prev_token] , ctx ]  : [32 x 1024]
// ---------------------------------------------------------------------------
__global__ void stage_rnn_in(const int* __restrict__ trg, const float* __restrict__ embed,
                             const float* __restrict__ ctx, _Float16* __restrict__ rnn_in, int t) {
    const int i = blockIdx.x * 256 + threadIdx.x;   // 32*1024
    const int b = i >> 10, c = i & 1023;
    float v;
    if (c < Ee) {
        const int tok = (t == 0) ? 1 : trg[b * Tt + t - 1];
        v = embed[(size_t)tok * Ee + c];
    } else {
        v = ctx[(size_t)b * Hh + (c - Ee)];
    }
    rnn_in[i] = (_Float16)v;
}

// ---------------------------------------------------------------------------
// GRU layer-0 nonlinearity (gate order r,z,n); updates h0 f32 + f16.
// ---------------------------------------------------------------------------
__global__ void gru0_update(const float* __restrict__ gi, const float* __restrict__ gh,
                            float* __restrict__ h, _Float16* __restrict__ hh) {
    const int i = blockIdx.x * 256 + threadIdx.x;   // 32*512
    const int b = i >> 9, j = i & 511;
    const float* gib = gi + (size_t)b * G3H;
    const float* ghb = gh + (size_t)b * G3H;
    const float r = sigmoidf_(gib[j] + ghb[j]);
    const float z = sigmoidf_(gib[Hh + j] + ghb[Hh + j]);
    const float n = tanhf(gib[2 * Hh + j] + r * ghb[2 * Hh + j]);
    const float hn = (1.0f - z) * n + z * h[i];
    h[i] = hn;
    hh[i] = (_Float16)hn;
}

// ---------------------------------------------------------------------------
// GRU layer-1 + attention + ctxn.  One block per batch row.
// ---------------------------------------------------------------------------
__global__ void attn_kernel(const float* __restrict__ gi, const float* __restrict__ gh,
                            float* __restrict__ h1, _Float16* __restrict__ h1h,
                            const float* __restrict__ proj_key, const float* __restrict__ attr_cat,
                            const float* __restrict__ query_w, const float* __restrict__ energy_w,
                            const float* __restrict__ ctx_w, const float* __restrict__ ctx_b,
                            float* __restrict__ ctx, _Float16* __restrict__ ctxnh) {
    __shared__ float sh_vec[Hh + Aa];   // [0:512]=h1n  [512:1024]=context
    __shared__ float sh_q[Hh];
    __shared__ float red[256];
    __shared__ float sh_alpha[3];
    const int b = blockIdx.x, tid = threadIdx.x;

    // layer-1 GRU cell
    for (int j = tid; j < Hh; j += 256) {
        const float r = sigmoidf_(gi[(size_t)b * G3H + j] + gh[(size_t)b * G3H + j]);
        const float z = sigmoidf_(gi[(size_t)b * G3H + Hh + j] + gh[(size_t)b * G3H + Hh + j]);
        const float n = tanhf(gi[(size_t)b * G3H + 2 * Hh + j] + r * gh[(size_t)b * G3H + 2 * Hh + j]);
        const float hn = (1.0f - z) * n + z * h1[(size_t)b * Hh + j];
        sh_vec[j] = hn;
        h1[(size_t)b * Hh + j] = hn;
        h1h[(size_t)b * Hh + j] = (_Float16)hn;
    }
    __syncthreads();
    // q = h1n @ query_w.T
    for (int h = tid; h < Hh; h += 256) {
        const float* qr = query_w + (size_t)h * Hh;
        float a = 0.0f;
        for (int k = 0; k < Hh; ++k) a = fmaf(sh_vec[k], qr[k], a);
        sh_q[h] = a;
    }
    __syncthreads();
    // scores[s] = sum_h tanh(q[h] + proj_key[b,s,h]) * energy_w[h]; softmax over 3
    float sc[3];
    for (int s = 0; s < 3; ++s) {
        float p = 0.0f;
        const float* pk = proj_key + (size_t)b * (3 * Hh) + s * Hh;
        for (int h = tid; h < Hh; h += 256) p += tanhf(sh_q[h] + pk[h]) * energy_w[h];
        red[tid] = p; __syncthreads();
        for (int off = 128; off > 0; off >>= 1) {
            if (tid < off) red[tid] += red[tid + off];
            __syncthreads();
        }
        sc[s] = red[0]; __syncthreads();
    }
    if (tid == 0) {
        float mx = fmaxf(sc[0], fmaxf(sc[1], sc[2]));
        float e0 = __expf(sc[0] - mx), e1 = __expf(sc[1] - mx), e2 = __expf(sc[2] - mx);
        float inv = 1.0f / (e0 + e1 + e2);
        sh_alpha[0] = e0 * inv; sh_alpha[1] = e1 * inv; sh_alpha[2] = e2 * inv;
    }
    __syncthreads();
    // context = sum_s alpha[s] * encoder_attr[b,s,:]
    for (int a = tid; a < Aa; a += 256) {
        const float* cb = attr_cat + (size_t)b * CAT;
        sh_vec[Hh + a] = sh_alpha[0] * cb[a] + sh_alpha[1] * cb[Aa + a] + sh_alpha[2] * cb[2 * Aa + a];
    }
    __syncthreads();
    // ctxn = tanh([h1n, context] @ ctx_w.T + ctx_b); becomes next ctx + generator input
    for (int h = tid; h < Hh; h += 256) {
        const float* wr = ctx_w + (size_t)h * (Hh + Aa);
        float a = ctx_b[h];
        for (int k = 0; k < Hh + Aa; ++k) a = fmaf(sh_vec[k], wr[k], a);
        const float c = tanhf(a);
        ctx[(size_t)b * Hh + h] = c;
        ctxnh[(size_t)b * Hh + h] = (_Float16)c;
    }
}

// ---------------------------------------------------------------------------
// Row log-softmax over V=30000, write out[b, t, :].  One block per batch row.
// ---------------------------------------------------------------------------
__global__ void log_softmax_out(const float* __restrict__ logits, float* __restrict__ out, int t) {
    __shared__ float red[512];
    const int b = blockIdx.x, tid = threadIdx.x;
    const float* row = logits + (size_t)b * Vv;
    float mx = -3.4e38f;
    for (int v = tid; v < Vv; v += 512) mx = fmaxf(mx, row[v]);
    red[tid] = mx; __syncthreads();
    for (int off = 256; off > 0; off >>= 1) {
        if (tid < off) red[tid] = fmaxf(red[tid], red[tid + off]);
        __syncthreads();
    }
    mx = red[0]; __syncthreads();
    float s = 0.0f;
    for (int v = tid; v < Vv; v += 512) s += __expf(row[v] - mx);
    red[tid] = s; __syncthreads();
    for (int off = 256; off > 0; off >>= 1) {
        if (tid < off) red[tid] += red[tid + off];
        __syncthreads();
    }
    const float ls = __logf(red[0]);
    float* orow = out + ((size_t)b * Tt + t) * Vv;
    for (int v = tid; v < Vv; v += 512) orow[v] = row[v] - mx - ls;
}

// ---------------------------------------------------------------------------
extern "C" void kernel_launch(void* const* d_in, const int* in_sizes, int n_in,
                              void* d_out, int out_size, void* d_ws, size_t ws_size,
                              hipStream_t stream) {
    (void)in_sizes; (void)n_in; (void)out_size; (void)ws_size;
    const int*   user      = (const int*)d_in[0];
    const int*   product   = (const int*)d_in[1];
    const int*   rating    = (const int*)d_in[2];
    const int*   trg       = (const int*)d_in[3];
    const float* embed     = (const float*)d_in[5];
    const float* user_w    = (const float*)d_in[6];
    const float* product_w = (const float*)d_in[7];
    const float* rating_w  = (const float*)d_in[8];
    const float* enc_w     = (const float*)d_in[9];
    const float* enc_b     = (const float*)d_in[10];
    const float* w_ih0     = (const float*)d_in[11];
    const float* w_hh0     = (const float*)d_in[12];
    const float* b_ih0     = (const float*)d_in[13];
    const float* b_hh0     = (const float*)d_in[14];
    const float* w_ih1     = (const float*)d_in[15];
    const float* w_hh1     = (const float*)d_in[16];
    const float* b_ih1     = (const float*)d_in[17];
    const float* b_hh1     = (const float*)d_in[18];
    const float* key_w     = (const float*)d_in[19];
    const float* query_w   = (const float*)d_in[20];
    const float* energy_w  = (const float*)d_in[21];
    const float* ctx_w     = (const float*)d_in[22];
    const float* ctx_b     = (const float*)d_in[23];
    const float* gen_w     = (const float*)d_in[24];
    float* out = (float*)d_out;

    // ---- workspace bump allocator (256B aligned) ----
    char* p = (char*)d_ws;
    auto alloc = [&](size_t bytes) -> void* {
        void* r = (void*)p;
        p += (bytes + 255) & ~(size_t)255;
        return r;
    };
    _Float16* gen_w_h  = (_Float16*)alloc((size_t)Vv * Hh * 2);
    _Float16* wih0_h   = (_Float16*)alloc((size_t)G3H * (Ee + Hh) * 2);
    _Float16* whh0_h   = (_Float16*)alloc((size_t)G3H * Hh * 2);
    _Float16* wih1_h   = (_Float16*)alloc((size_t)G3H * Hh * 2);
    _Float16* whh1_h   = (_Float16*)alloc((size_t)G3H * Hh * 2);
    float*    logits   = (float*)   alloc((size_t)Bb * Vv * 4);
    float*    gi0      = (float*)   alloc((size_t)Bb * G3H * 4);
    float*    gh0      = (float*)   alloc((size_t)Bb * G3H * 4);
    float*    gi1      = (float*)   alloc((size_t)Bb * G3H * 4);
    float*    gh1      = (float*)   alloc((size_t)Bb * G3H * 4);
    _Float16* rnn_in_h = (_Float16*)alloc((size_t)Bb * (Ee + Hh) * 2);
    float*    h0       = (float*)   alloc((size_t)Bb * Hh * 4);
    _Float16* h0h      = (_Float16*)alloc((size_t)Bb * Hh * 2);
    float*    h1       = (float*)   alloc((size_t)Bb * Hh * 4);
    _Float16* h1h      = (_Float16*)alloc((size_t)Bb * Hh * 2);
    float*    ctx      = (float*)   alloc((size_t)Bb * Hh * 4);
    _Float16* ctxnh    = (_Float16*)alloc((size_t)Bb * Hh * 2);
    float*    attr_cat = (float*)   alloc((size_t)Bb * CAT * 4);
    float*    proj_key = (float*)   alloc((size_t)Bb * 3 * Hh * 4);

    // ---- one-time f32->f16 weight conversion ----
    cvt_f16_kernel<<<2048, 256, 0, stream>>>(gen_w, gen_w_h, (long)Vv * Hh);
    cvt_f16_kernel<<<1024, 256, 0, stream>>>(w_ih0, wih0_h, (long)G3H * (Ee + Hh));
    cvt_f16_kernel<<<512, 256, 0, stream>>>(w_hh0, whh0_h, (long)G3H * Hh);
    cvt_f16_kernel<<<512, 256, 0, stream>>>(w_ih1, wih1_h, (long)G3H * Hh);
    cvt_f16_kernel<<<512, 256, 0, stream>>>(w_hh1, whh1_h, (long)G3H * Hh);

    // ---- encoder / initial states ----
    encoder_kernel<<<Bb, 256, 0, stream>>>(user, product, rating, user_w, product_w, rating_w,
                                           enc_w, enc_b, key_w,
                                           attr_cat, proj_key, h0, h0h, h1, h1h, ctx);

    // ---- sequential decode ----
    const int NT_G = Vv / 16;            // 1875 n-tiles for generator
    const int NT_H = G3H / 16;           // 96 n-tiles for gates
    for (int t = 0; t < Tt; ++t) {
        stage_rnn_in<<<(Bb * (Ee + Hh)) / 256, 256, 0, stream>>>(trg, embed, ctx, rnn_in_h, t);
        gemm_xwT_wmma<<<(NT_H + 3) / 4, 256, 0, stream>>>(rnn_in_h, wih0_h, b_ih0, gi0, Ee + Hh, NT_H);
        gemm_xwT_wmma<<<(NT_H + 3) / 4, 256, 0, stream>>>(h0h, whh0_h, b_hh0, gh0, Hh, NT_H);
        gru0_update<<<(Bb * Hh) / 256, 256, 0, stream>>>(gi0, gh0, h0, h0h);
        gemm_xwT_wmma<<<(NT_H + 3) / 4, 256, 0, stream>>>(h0h, wih1_h, b_ih1, gi1, Hh, NT_H);
        gemm_xwT_wmma<<<(NT_H + 3) / 4, 256, 0, stream>>>(h1h, whh1_h, b_hh1, gh1, Hh, NT_H);
        attn_kernel<<<Bb, 256, 0, stream>>>(gi1, gh1, h1, h1h, proj_key, attr_cat,
                                            query_w, energy_w, ctx_w, ctx_b, ctx, ctxnh);
        gemm_xwT_wmma<<<(NT_G + 3) / 4, 256, 0, stream>>>(ctxnh, gen_w_h, nullptr, logits, Hh, NT_G);
        log_softmax_out<<<Bb, 512, 0, stream>>>(logits, out, t);
    }
}
